// ESCMLayer_14731737825693
// MI455X (gfx1250) — matrix operand order repaired and checked by reading, not compiled, not run on hardware
//
#include <hip/hip_runtime.h>
#include <hip/hip_bf16.h>
#include <math.h>

typedef __attribute__((ext_vector_type(16))) _Float16 v16h;
typedef __attribute__((ext_vector_type(8)))  _Float16 v8h;
typedef __attribute__((ext_vector_type(8)))  float    v8f;
typedef __attribute__((ext_vector_type(4)))  float    v4f;

#define BATCH   16384
#define NFIELD  10
#define EMBD    16
#define LPF     20
#define FEAT    160      // NFIELD * EMBD
#define NEXP    8
#define SDIM    64
#define NGATE   3
#define TDIM    32
#define EPSV    1e-15f

// workspace layout (bytes)
#define X_OFF   0                              // float   [BATCH][FEAT]   10.5 MB
#define XH_OFF  (BATCH * FEAT * 4)             // f16     [BATCH][FEAT]    5.2 MB
#define WHT_OFF (XH_OFF + BATCH * FEAT * 2)    // f16     [E][S][FEAT]   160 KB (K-contiguous)

// ---------------------------------------------------------------------------
// Kernel 1: transpose expert weights (E,FEAT,S) f32 -> (E,S,FEAT) f16 so each
// WMMA B fragment is two contiguous global_load_b128 per lane.
// ---------------------------------------------------------------------------
__global__ void repack_w_kernel(const float* __restrict__ ew,
                                _Float16* __restrict__ whT) {
    int idx = blockIdx.x * 256 + threadIdx.x;
    if (idx >= NEXP * SDIM * FEAT) return;
    int e = idx / (SDIM * FEAT);
    int r = idx % (SDIM * FEAT);
    int n = r / FEAT;
    int k = r % FEAT;
    whT[idx] = (_Float16)ew[(e * FEAT + k) * SDIM + n];
}

// ---------------------------------------------------------------------------
// Kernel 2: embedding gather + sum over L.  One block per batch row,
// thread = f*16+d, 16 lanes of a field read consecutive dims (coalesced 64B
// per id).  Writes x (f32, for VALU tail) and xh (f16, WMMA A operand).
// ---------------------------------------------------------------------------
__global__ void gather_embed_kernel(const long long* __restrict__ ids,
                                    const float* __restrict__ emb,
                                    float* __restrict__ x,
                                    _Float16* __restrict__ xh) {
    const int b = blockIdx.x;
    const int K = threadIdx.x;          // 0..159
    const int f = K >> 4;
    const int d = K & 15;
    const long long* idp = ids + ((long long)f * BATCH + b) * LPF;
    float s = 0.f;
#pragma unroll
    for (int l = 0; l < LPF; ++l) {
        long long id = idp[l];
        s += emb[(size_t)id * EMBD + d];
    }
    x[(size_t)b * FEAT + K]  = s;
    xh[(size_t)b * FEAT + K] = (_Float16)s;
}

// ---------------------------------------------------------------------------
// Kernel 3: fused MMoE.  64 threads = 2 waves per block, each wave owns one
// 16-row tile.  Phase A: experts via v_wmma_f32_16x16x32_f16 (5 K-chunks x
// 8 experts x 4 N-tiles), bias+ReLU, scatter to LDS in row-major order.
// Phase B (lanes 0..15): gates softmax, mix, tower, heads, output assembly.
// ---------------------------------------------------------------------------
__global__ __launch_bounds__(64)
void mmoe_kernel(const _Float16* __restrict__ xh,
                 const float* __restrict__ x,
                 const _Float16* __restrict__ whT,
                 const float* __restrict__ expert_b,
                 const float* __restrict__ gate_w,
                 const float* __restrict__ gate_b,
                 const float* __restrict__ tower_w,
                 const float* __restrict__ tower_b,
                 const float* __restrict__ out_w,
                 const float* __restrict__ out_b,
                 float* __restrict__ out) {
    __shared__ float lds_exp[2][16][NEXP * SDIM];   // 64 KB

    const int tid  = threadIdx.x;
    const int w    = tid >> 5;
    const int lane = tid & 31;
    const int M    = lane & 15;
    const int h    = lane >> 4;
    const int tile = blockIdx.x * 2 + w;

    // ---- Phase A: expert GEMM on the WMMA pipe -------------------------
    // A fragment (16-bit A 16x32, ISA 7.12.2): lane M holds
    //   elems 0..7  -> K = K0 + 8h + e        (contiguous 16B)
    //   elems 8..15 -> K = K0 + 16 + 8h + e   (contiguous 16B)
    v16h afr[5];
    {
        const _Float16* xr = xh + (size_t)(tile * 16 + M) * FEAT;
#pragma unroll
        for (int kc = 0; kc < 5; ++kc) {
            union { v16h v; struct { v8h lo; v8h hi; } s; } u;
            u.s.lo = *(const v8h*)(xr + kc * 32 + 8 * h);
            u.s.hi = *(const v8h*)(xr + kc * 32 + 16 + 8 * h);
            afr[kc] = u.v;
        }
    }

    for (int e = 0; e < NEXP; ++e) {
#pragma unroll
        for (int nt = 0; nt < 4; ++nt) {
            v8f acc = {};
            // B fragment (16-bit B 32x16): lane N holds K = K0+8h .. +7 and
            // K0+16+8h .. +7, from K-major whT -> two 16B loads.
            const _Float16* wb =
                whT + ((size_t)e * SDIM + nt * 16 + M) * FEAT + 8 * h;
#pragma unroll
            for (int kc = 0; kc < 5; ++kc) {
                union { v16h v; struct { v8h lo; v8h hi; } s; } u;
                u.s.lo = *(const v8h*)(wb + kc * 32);
                u.s.hi = *(const v8h*)(wb + kc * 32 + 16);
                acc = __builtin_amdgcn_wmma_f32_16x16x32_f16(
                    false, afr[kc], false, u.v, (short)0, acc, false, false);
            }
            // C layout: VGPR r, lanes 0-15 -> row r, lanes 16-31 -> row 8+r,
            // column N = nt*16 + (lane&15).  Bias depends only on column.
            const int col = nt * 16 + M;
            const float bias = expert_b[e * SDIM + col];
#pragma unroll
            for (int r = 0; r < 8; ++r) {
                float v = acc[r] + bias;
                v = v > 0.f ? v : 0.f;                       // ReLU
                lds_exp[w][8 * h + r][e * SDIM + col] = v;
            }
        }
    }
    __syncthreads();

    // ---- Phase B: per-row tail (gates / mix / tower / heads) -----------
    if (lane < 16) {
        const int brow = tile * 16 + lane;
        const float* xrow = x + (size_t)brow * FEAT;

        // gates: (160 -> 3x8) + softmax over experts
        float g[NGATE][NEXP];
#pragma unroll
        for (int gg = 0; gg < NGATE; ++gg)
#pragma unroll
            for (int e = 0; e < NEXP; ++e)
                g[gg][e] = gate_b[gg * NEXP + e];

        for (int k = 0; k < FEAT; ++k) {
            const float xv = xrow[k];
#pragma unroll
            for (int gg = 0; gg < NGATE; ++gg)
#pragma unroll
                for (int e = 0; e < NEXP; ++e)
                    g[gg][e] += xv * gate_w[(gg * FEAT + k) * NEXP + e];
        }
#pragma unroll
        for (int gg = 0; gg < NGATE; ++gg) {
            float mx = g[gg][0];
#pragma unroll
            for (int e = 1; e < NEXP; ++e) mx = fmaxf(mx, g[gg][e]);
            float sum = 0.f;
#pragma unroll
            for (int e = 0; e < NEXP; ++e) {
                g[gg][e] = expf(g[gg][e] - mx);
                sum += g[gg][e];
            }
            const float inv = 1.f / sum;
#pragma unroll
            for (int e = 0; e < NEXP; ++e) g[gg][e] *= inv;
        }

        // per-gate: mix -> tower(ReLU) -> 2-way head softmax (clipped)
        float pv0[NGATE], pv1[NGATE];
        for (int gg = 0; gg < NGATE; ++gg) {
            float tow[TDIM];
#pragma unroll
            for (int t = 0; t < TDIM; ++t) tow[t] = tower_b[gg * TDIM + t];

            for (int sdx = 0; sdx < SDIM; ++sdx) {
                float m = 0.f;
#pragma unroll
                for (int e = 0; e < NEXP; ++e)
                    m += g[gg][e] * lds_exp[w][lane][e * SDIM + sdx];
                const float* tw = tower_w + (size_t)(gg * SDIM + sdx) * TDIM;
#pragma unroll
                for (int t = 0; t < TDIM; ++t) tow[t] += m * tw[t];
            }
            float l0 = out_b[gg * 2 + 0];
            float l1 = out_b[gg * 2 + 1];
#pragma unroll
            for (int t = 0; t < TDIM; ++t) {
                const float tv = tow[t] > 0.f ? tow[t] : 0.f;    // ReLU
                l0 += tv * out_w[(gg * TDIM + t) * 2 + 0];
                l1 += tv * out_w[(gg * TDIM + t) * 2 + 1];
            }
            const float mx = fmaxf(l0, l1);
            const float e0 = expf(l0 - mx), e1 = expf(l1 - mx);
            const float inv = 1.f / (e0 + e1);
            pv0[gg] = fminf(fmaxf(e0 * inv, EPSV), 1.f - EPSV);
            pv1[gg] = fminf(fmaxf(e1 * inv, EPSV), 1.f - EPSV);
        }

        // output: [ctr0,ctr1, ctr1, cvr0,cvr1, cvr1, 1-cc, cc, cc, imp1]
        const float ctr1 = pv1[0], cvr1 = pv1[1];
        const float cc = ctr1 * cvr1;
        float* o = out + (size_t)brow * 10;
        o[0] = pv0[0]; o[1] = ctr1;    o[2] = ctr1;
        o[3] = pv0[1]; o[4] = cvr1;    o[5] = cvr1;
        o[6] = 1.f - cc; o[7] = cc;    o[8] = cc;
        o[9] = pv1[2];
    }
}

// ---------------------------------------------------------------------------
extern "C" void kernel_launch(void* const* d_in, const int* in_sizes, int n_in,
                              void* d_out, int out_size, void* d_ws, size_t ws_size,
                              hipStream_t stream) {
    const long long* ids      = (const long long*)d_in[0];  // int64 (F,B,L)
    const float*     emb      = (const float*)d_in[1];      // (V,16)
    const float*     expert_w = (const float*)d_in[2];      // (E,160,64)
    const float*     expert_b = (const float*)d_in[3];      // (E,64)
    const float*     gate_w   = (const float*)d_in[4];      // (G,160,E)
    const float*     gate_b   = (const float*)d_in[5];      // (G,E)
    const float*     tower_w  = (const float*)d_in[6];      // (G,64,32)
    const float*     tower_b  = (const float*)d_in[7];      // (G,32)
    const float*     out_w    = (const float*)d_in[8];      // (G,32,2)
    const float*     out_b    = (const float*)d_in[9];      // (G,2)
    float* out = (float*)d_out;

    char* ws = (char*)d_ws;
    float*    x   = (float*)(ws + X_OFF);
    _Float16* xh  = (_Float16*)(ws + XH_OFF);
    _Float16* whT = (_Float16*)(ws + WHT_OFF);

    repack_w_kernel<<<(NEXP * SDIM * FEAT + 255) / 256, 256, 0, stream>>>(
        expert_w, whT);
    gather_embed_kernel<<<BATCH, FEAT, 0, stream>>>(ids, emb, x, xh);
    mmoe_kernel<<<BATCH / 32, 64, 0, stream>>>(
        xh, x, whT, expert_b, gate_w, gate_b, tower_w, tower_b,
        out_w, out_b, out);
}
